// GroupContrastModel_5660766896673
// MI455X (gfx1250) — compile-verified, matrix operand order checked by reading, not run on hardware
//
#include <hip/hip_runtime.h>
#include <hip/hip_bf16.h>

typedef __attribute__((ext_vector_type(16))) _Float16 v16h;
typedef __attribute__((ext_vector_type(8)))  float    v8f;
typedef __attribute__((ext_vector_type(4), may_alias)) float f4;

#define NROWS 8192
#define KDIM  300
#define KP    320           // K padded to multiple of 32
#define NKF   10            // KP/32 WMMA k-steps
#define TILE_J 16
#define CSPLIT 16
#define JT_PER (NROWS / TILE_J / CSPLIT)   // 32 column tiles per block
#define MLP_ROWS 8

// ---------------- kernel A: zero the label histogram ----------------
__global__ void zero_kernel(int* gcnt) {
    if (threadIdx.x < 128) gcnt[threadIdx.x] = 0;
}

// ---------------- kernel B: label histogram ----------------
__global__ void hist_kernel(const int* __restrict__ labels, int* __restrict__ gcnt) {
    int i = blockIdx.x * blockDim.x + threadIdx.x;
    if (i < NROWS) atomicAdd(&gcnt[labels[i]], 1);
}

// ---------------- kernel C: MLP + normalize + f16 cast + diag self-dot ----------------
__global__ __launch_bounds__(320) void mlp_kernel(
    const float* __restrict__ x, const float* __restrict__ W1, const float* __restrict__ b1,
    const float* __restrict__ W2, const float* __restrict__ b2,
    _Float16* __restrict__ F, float* __restrict__ diag)
{
    __shared__ float xs[MLP_ROWS][KDIM + 4];
    __shared__ float hs[MLP_ROWS][KDIM + 4];
    __shared__ float nacc[MLP_ROWS];
    __shared__ float dacc[MLP_ROWS];
    const int tid  = threadIdx.x;
    const int row0 = blockIdx.x * MLP_ROWS;

    for (int idx = tid; idx < MLP_ROWS * KDIM; idx += 320) {
        int r = idx / KDIM, c = idx % KDIM;
        xs[r][c] = x[(size_t)(row0 + r) * KDIM + c];
    }
    if (tid < MLP_ROWS) { nacc[tid] = 0.0f; dacc[tid] = 0.0f; }
    __syncthreads();

    float acc[MLP_ROWS];
    if (tid < KDIM) {
        for (int r = 0; r < MLP_ROWS; ++r) acc[r] = b1[tid];
        for (int k = 0; k < KDIM; ++k) {
            float w = W1[(size_t)k * KDIM + tid];
#pragma unroll
            for (int r = 0; r < MLP_ROWS; ++r) acc[r] = fmaf(xs[r][k], w, acc[r]);
        }
        for (int r = 0; r < MLP_ROWS; ++r) hs[r][tid] = fmaxf(acc[r], 0.0f);
    }
    __syncthreads();

    if (tid < KDIM) {
        for (int r = 0; r < MLP_ROWS; ++r) acc[r] = b2[tid];
        for (int k = 0; k < KDIM; ++k) {
            float w = W2[(size_t)k * KDIM + tid];
#pragma unroll
            for (int r = 0; r < MLP_ROWS; ++r) acc[r] = fmaf(hs[r][k], w, acc[r]);
        }
        for (int r = 0; r < MLP_ROWS; ++r) atomicAdd(&nacc[r], acc[r] * acc[r]);
    }
    __syncthreads();

    if (tid < KDIM) {
        for (int r = 0; r < MLP_ROWS; ++r) {
            float rn = rsqrtf(nacc[r]);
            _Float16 fh = (_Float16)(acc[r] * rn);
            F[(size_t)(row0 + r) * KP + tid] = fh;
            float ff = (float)fh;
            atomicAdd(&dacc[r], ff * ff);
        }
    } else if (tid < KP) {
        for (int r = 0; r < MLP_ROWS; ++r)
            F[(size_t)(row0 + r) * KP + tid] = (_Float16)0.0f;
    }
    __syncthreads();
    if (tid < MLP_ROWS) diag[row0 + tid] = dacc[tid];
}

// ---------------- kernel D: fused f@f^T WMMA + streaming contrastive reductions ----------------
__global__ __launch_bounds__(256) void contrast_kernel(
    const _Float16* __restrict__ F, const float* __restrict__ diag,
    const int* __restrict__ labels, const int* __restrict__ gcnt,
    float* __restrict__ Zp, float* __restrict__ Sp)
{
    __shared__ _Float16 Bs[2][TILE_J][KP + 8];   // +8 halfs pad -> conflict-free strided reads
    __shared__ int labT[2][TILE_J];

    const int tid  = threadIdx.x;
    const int wave = tid >> 5;
    const int lane = tid & 31;
    const int hi   = lane >> 4;       // which half of the wave
    const int ln16 = lane & 15;
    const int r0   = blockIdx.x * 128 + wave * 16;   // 16 rows owned by this wave
    const int cs   = blockIdx.y;                     // column split

    // Resident A fragments: 16 rows x 320 halfs, WMMA 16-bit A layout:
    // lane half 'hi' holds K chunks [kk*32 + hi*8 .. +7] and [kk*32 + 16 + hi*8 .. +7]
    v16h a[NKF];
    {
        const _Float16* rp = F + (size_t)(r0 + ln16) * KP;
#pragma unroll
        for (int kk = 0; kk < NKF; ++kk) {
            f4* d = (f4*)&a[kk];
            d[0] = *(const f4*)(rp + kk * 32 + hi * 8);
            d[1] = *(const f4*)(rp + kk * 32 + hi * 8 + 16);
        }
    }

    // Per-row folded constants for the 8 rows this lane accumulates (C/D layout: M = hi*8 + v)
    // e = dot*invT - wsel   where wsel = diag*invT + log_pos (same) / + log_neg (diff)
    const float invT = 100.0f;   // 1 / TEMPERATURE
    int   labi[8], rowi[8];
    float wsame[8], wdiff[8];
#pragma unroll
    for (int v = 0; v < 8; ++v) {
        int i = r0 + hi * 8 + v;
        rowi[v] = i;
        labi[v] = labels[i];
        float dT = diag[i] * invT;
        float pc = (float)(gcnt[labi[v]] - 1);
        float ng = (float)(NROWS - 1) - pc;
        wsame[v] = dT + (pc > 0.0f ? __logf(pc) : 0.0f);
        wdiff[v] = dT + (ng > 0.0f ? __logf(ng) : 0.0f);
    }
    float Z[8], S[8];
#pragma unroll
    for (int v = 0; v < 8; ++v) { Z[v] = 0.0f; S[v] = 0.0f; }

    const int jt0 = cs * JT_PER;

    // prologue: stage first B tile into buffer 0
    {
        int j0 = jt0 * TILE_J;
        for (int idx = tid; idx < TILE_J * (KP / 8); idx += 256) {
            int r = idx / (KP / 8), c = idx % (KP / 8);
            *(f4*)&Bs[0][r][c * 8] = *(const f4*)(F + (size_t)(j0 + r) * KP + c * 8);
        }
        if (tid < TILE_J) labT[0][tid] = labels[j0 + tid];
    }

    int cur = 0;
    for (int t = 0; t < JT_PER; ++t) {
        __syncthreads();
        if (t + 1 < JT_PER) {            // stage next tile into the other buffer
            int j0 = (jt0 + t + 1) * TILE_J;
            int nb = cur ^ 1;
            for (int idx = tid; idx < TILE_J * (KP / 8); idx += 256) {
                int r = idx / (KP / 8), c = idx % (KP / 8);
                *(f4*)&Bs[nb][r][c * 8] = *(const f4*)(F + (size_t)(j0 + r) * KP + c * 8);
            }
            if (tid < TILE_J) labT[nb][tid] = labels[j0 + tid];
        }

        // 16x16 logits tile via 10 chained WMMAs (f16 in, f32 accumulate)
        v8f c8 = {};
#pragma unroll
        for (int kk = 0; kk < NKF; ++kk) {
            v16h bv;
            const _Float16* bp = &Bs[cur][ln16][kk * 32 + hi * 8];
            f4* d = (f4*)&bv;
            d[0] = *(const f4*)bp;
            d[1] = *(const f4*)(bp + 16);
            c8 = __builtin_amdgcn_wmma_f32_16x16x32_f16(
                     false, a[kk], false, bv, (short)0, c8, false, false);
        }

        // streaming reductions, fully branchless (cndmask + fma, no exec divergence)
        int j    = (jt0 + t) * TILE_J + ln16;   // C/D layout: N = lane%16
        int labj = labT[cur][ln16];
#pragma unroll
        for (int v = 0; v < 8; ++v) {
            bool same   = (labj == labi[v]);
            float wsel  = same ? wsame[v] : wdiff[v];
            float e     = fmaf(c8[v], invT, -wsel);
            float ex    = __expf(e);
            float mdiag = (j != rowi[v]) ? 1.0f : 0.0f;
            float mpos  = same ? mdiag : 0.0f;
            Z[v] = fmaf(mdiag, ex, Z[v]);
            S[v] = fmaf(mpos,  e,  S[v]);
        }
        cur ^= 1;
    }

    // reduce Z,S across the 16 lanes of each half (they share the same 8 rows)
#pragma unroll
    for (int m = 1; m < 16; m <<= 1) {
#pragma unroll
        for (int v = 0; v < 8; ++v) {
            Z[v] += __shfl_xor(Z[v], m, 32);
            S[v] += __shfl_xor(S[v], m, 32);
        }
    }
    if (ln16 == 0) {
#pragma unroll
        for (int v = 0; v < 8; ++v) {
            int i = rowi[v];
            Zp[(size_t)cs * NROWS + i] = Z[v];
            Sp[(size_t)cs * NROWS + i] = S[v];
        }
    }
}

// ---------------- kernel E1: combine column splits, per-row mean log-prob ----------------
__global__ void rowfin_kernel(const float* __restrict__ Zp, const float* __restrict__ Sp,
                              const int* __restrict__ labels, const int* __restrict__ gcnt,
                              float* __restrict__ mlpv, float* __restrict__ validf)
{
    int i = blockIdx.x * blockDim.x + threadIdx.x;
    if (i >= NROWS) return;
    float Z = 0.0f, S = 0.0f;
#pragma unroll
    for (int c = 0; c < CSPLIT; ++c) {
        Z += Zp[(size_t)c * NROWS + i];
        S += Sp[(size_t)c * NROWS + i];
    }
    float pc = (float)(gcnt[labels[i]] - 1);
    if (pc > 0.0f) {
        mlpv[i]   = (S - pc * __logf(Z)) / pc;
        validf[i] = 1.0f;
    } else {
        mlpv[i]   = 0.0f;
        validf[i] = 0.0f;
    }
}

// ---------------- kernel E2: final scalar loss ----------------
__global__ __launch_bounds__(256) void loss_kernel(const float* __restrict__ mlpv,
                                                   const float* __restrict__ validf,
                                                   float* __restrict__ out)
{
    __shared__ float sn[256], sv[256];
    int tid = threadIdx.x;
    float n = 0.0f, v = 0.0f;
    for (int i = tid; i < NROWS; i += 256) { n += mlpv[i]; v += validf[i]; }
    sn[tid] = n; sv[tid] = v;
    __syncthreads();
    for (int s = 128; s > 0; s >>= 1) {
        if (tid < s) { sn[tid] += sn[tid + s]; sv[tid] += sv[tid + s]; }
        __syncthreads();
    }
    if (tid == 0) out[0] = -0.01f * sn[0] / sv[0];
}

extern "C" void kernel_launch(void* const* d_in, const int* in_sizes, int n_in,
                              void* d_out, int out_size, void* d_ws, size_t ws_size,
                              hipStream_t stream) {
    (void)in_sizes; (void)n_in; (void)out_size; (void)ws_size;
    const float* x  = (const float*)d_in[0];
    const float* W1 = (const float*)d_in[1];
    const float* b1 = (const float*)d_in[2];
    const float* W2 = (const float*)d_in[3];
    const float* b2 = (const float*)d_in[4];
    const int* labels = (const int*)d_in[5];

    char* ws = (char*)d_ws;
    _Float16* F   = (_Float16*)ws;              // 8192*320*2 = 5,242,880 B
    float* diag   = (float*)(ws + 5242880);     // 32 KB
    float* Zp     = (float*)(ws + 5275648);     // 16*8192*4 = 512 KB
    float* Sp     = (float*)(ws + 5799936);     // 512 KB
    float* mlpv   = (float*)(ws + 6324224);     // 32 KB
    float* validf = (float*)(ws + 6356992);     // 32 KB
    int*   gcnt   = (int*)(ws + 6389760);       // 128 ints

    zero_kernel<<<1, 128, 0, stream>>>(gcnt);
    hist_kernel<<<NROWS / 256, 256, 0, stream>>>(labels, gcnt);
    mlp_kernel<<<NROWS / MLP_ROWS, 320, 0, stream>>>(x, W1, b1, W2, b2, F, diag);
    contrast_kernel<<<dim3(NROWS / 128, CSPLIT), 256, 0, stream>>>(F, diag, labels, gcnt, Zp, Sp);
    rowfin_kernel<<<NROWS / 256, 256, 0, stream>>>(Zp, Sp, labels, gcnt, mlpv, validf);
    loss_kernel<<<1, 256, 0, stream>>>(mlpv, validf, (float*)d_out);
}
